// MoELayer_1468878815864
// MI455X (gfx1250) — matrix operand order repaired and checked by reading, not compiled, not run on hardware
//
#include <hip/hip_runtime.h>
#include <cstdint>
#include <cstddef>

// ---------------- problem constants ----------------
#define T_TOK   8192              // S*B
#define DM      1024              // D_MODEL
#define DF      4096              // D_FF
#define N_EXP   8

// ---------------- GEMM tiling ----------------
#define TM 128
#define TN 128
#define TKS 64                    // staged K per iteration (2 WMMA K-steps)
#define LDSW (TKS + 8)            // 72 bf16 = 144B row stride (16B multiple)
#define MAX_MT ((2 * T_TOK) / TM + N_EXP)   // 136 row tiles worst case
#define MAX_ROWS (MAX_MT * TM)              // 17408 padded slots

typedef __bf16 bf16;
typedef __attribute__((ext_vector_type(16))) __bf16 v16bf;
typedef __attribute__((ext_vector_type(8)))  __bf16 v8bf;
typedef __attribute__((ext_vector_type(8)))  float  v8f;

__device__ __forceinline__ unsigned short f2bf_bits(float f) {
  unsigned u = __builtin_bit_cast(unsigned, f);
  u += 0x7FFFu + ((u >> 16) & 1u);           // round-to-nearest-even
  return (unsigned short)(u >> 16);
}
__device__ __forceinline__ bf16 f2bf(float f) {
  unsigned short h = f2bf_bits(f);
  return __builtin_bit_cast(bf16, h);
}
__device__ __forceinline__ v16bf concat8(v8bf lo, v8bf hi) {
  return __builtin_shufflevector(lo, hi, 0,1,2,3,4,5,6,7,8,9,10,11,12,13,14,15);
}

// ---- CDNA5 async global->LDS (ASYNCcnt-tracked DMA, no VGPR round trip) ----
__device__ __forceinline__ void async_ld_b128(unsigned lds_off, const void* g) {
  // per-lane: LDS[lds_off] <- 16B at g
  asm volatile("global_load_async_to_lds_b128 %0, %1, off"
               :: "v"(lds_off), "v"(g) : "memory");
}
__device__ __forceinline__ void wait_async0() {
#if __has_builtin(__builtin_amdgcn_s_wait_asynccnt)
  __builtin_amdgcn_s_wait_asynccnt(0);
#else
  asm volatile("s_wait_asynccnt 0x0" ::: "memory");
#endif
}

// ---------------------------------------------------------------------------
// K0: transpose + f32->bf16 convert.  src: [E][R][C] f32, dst: [E][C][R] bf16
// (weights become N-major / K-contiguous so WMMA B-fragments read contiguous K)
// ---------------------------------------------------------------------------
template <int R, int C>
__global__ __launch_bounds__(256) void transpose_convert(
    const float* __restrict__ src, bf16* __restrict__ dst) {
  __shared__ __align__(16) float tile[64][68];
  const int e = blockIdx.z;
  const float* s = src + (size_t)e * R * C;
  bf16* d = dst + (size_t)e * R * C;
  const int c0 = blockIdx.x * 64;
  const int r0 = blockIdx.y * 64;
  const int tx = threadIdx.x & 15;   // 16 groups of 4 cols
  const int ty = threadIdx.x >> 4;   // 16 rows per pass
#pragma unroll
  for (int it = 0; it < 4; ++it) {
    int r = ty + it * 16;
    float4 v = *(const float4*)(s + (size_t)(r0 + r) * C + c0 + tx * 4);
    *(float4*)&tile[r][tx * 4] = v;
  }
  __syncthreads();
#pragma unroll
  for (int it = 0; it < 4; ++it) {
    int c = ty + it * 16;            // output row (= source column)
    int r4 = tx * 4;                 // output col group (= source rows)
    unsigned short h0 = f2bf_bits(tile[r4 + 0][c]);
    unsigned short h1 = f2bf_bits(tile[r4 + 1][c]);
    unsigned short h2 = f2bf_bits(tile[r4 + 2][c]);
    unsigned short h3 = f2bf_bits(tile[r4 + 3][c]);
    uint2 pk;
    pk.x = (unsigned)h0 | ((unsigned)h1 << 16);
    pk.y = (unsigned)h2 | ((unsigned)h3 << 16);
    *(uint2*)(d + (size_t)(c0 + c) * R + r0 + r4) = pk;
  }
}

// ---------------------------------------------------------------------------
// K1: gating. One wave (32 lanes) per token: scores = x@gate_w + gate_b,
// top-2 (reference tie semantics), 2-way softmax. Also emits x as bf16.
// ---------------------------------------------------------------------------
__global__ __launch_bounds__(256) void gate_kernel(
    const float* __restrict__ x, const float* __restrict__ gate_w,
    const float* __restrict__ gate_b, bf16* __restrict__ xb,
    int* __restrict__ topk_idx, float* __restrict__ topk_prob) {
  const int wave = threadIdx.x >> 5;
  const int lane = threadIdx.x & 31;
  const int t = blockIdx.x * 8 + wave;
  const float* xr = x + (size_t)t * DM;
  bf16* xbr = xb + (size_t)t * DM;

  float acc[8];
#pragma unroll
  for (int j = 0; j < 8; ++j) acc[j] = 0.0f;

  for (int i = lane; i < DM; i += 32) {
    float xv = xr[i];
    xbr[i] = f2bf(xv);
    const float4* g = (const float4*)(gate_w + (size_t)i * N_EXP);
    float4 g0 = g[0], g1 = g[1];
    acc[0] += xv * g0.x; acc[1] += xv * g0.y;
    acc[2] += xv * g0.z; acc[3] += xv * g0.w;
    acc[4] += xv * g1.x; acc[5] += xv * g1.y;
    acc[6] += xv * g1.z; acc[7] += xv * g1.w;
  }
#pragma unroll
  for (int j = 0; j < 8; ++j)
#pragma unroll
    for (int m = 16; m >= 1; m >>= 1) acc[j] += __shfl_xor(acc[j], m, 32);

  if (lane == 0) {
    float b1v = -3.4e38f, b2v = -3.4e38f;
    int i1 = 0, i2 = 0;
#pragma unroll
    for (int j = 0; j < 8; ++j) {
      float s = acc[j] + gate_b[j];
      if (s > b1v) { b2v = b1v; i2 = i1; b1v = s; i1 = j; }
      else if (s > b2v) { b2v = s; i2 = j; }
    }
    float e1 = expf(b2v - b1v);          // softmax over {b1v, b2v}
    float inv = 1.0f / (1.0f + e1);
    topk_idx[t * 2 + 0] = i1;  topk_idx[t * 2 + 1] = i2;
    topk_prob[t * 2 + 0] = inv; topk_prob[t * 2 + 1] = e1 * inv;
  }
}

// ---------------------------------------------------------------------------
// K2: routing. Single block: counts -> padded bases (multiples of TM) ->
// scatter slot assignments, tile->expert table, pad-row fill.
// meta[0]=numMTiles, meta[32+mt]=expert of tile mt.
// ---------------------------------------------------------------------------
__global__ __launch_bounds__(256) void route_kernel(
    const int* __restrict__ topk_idx, int* __restrict__ meta,
    int* __restrict__ tok_slot, int* __restrict__ row2tok) {
  __shared__ int cnt[N_EXP], off[N_EXP], base[N_EXP], pcnt[N_EXP];
  const int tid = threadIdx.x;
  if (tid < N_EXP) { cnt[tid] = 0; off[tid] = 0; }
  __syncthreads();
  for (int i = tid; i < 2 * T_TOK; i += 256) atomicAdd(&cnt[topk_idx[i]], 1);
  __syncthreads();
  if (tid == 0) {
    int b = 0, nt = 0;
    for (int e = 0; e < N_EXP; ++e) {
      base[e] = b;
      pcnt[e] = ((cnt[e] + TM - 1) / TM) * TM;
      int tiles = pcnt[e] / TM;
      for (int k = 0; k < tiles; ++k) meta[32 + nt + k] = e;
      nt += tiles;
      b += pcnt[e];
      meta[1 + e] = base[e];
    }
    for (int k = nt; k < MAX_MT; ++k) meta[32 + k] = 0;
    meta[0] = nt;
    meta[17] = b;
  }
  __syncthreads();
  for (int i = tid; i < 2 * T_TOK; i += 256) {
    int e = topk_idx[i];
    int s = base[e] + atomicAdd(&off[e], 1);
    tok_slot[i] = s;
    row2tok[s] = i >> 1;                 // token id
  }
  __syncthreads();
  for (int e = 0; e < N_EXP; ++e)        // pad rows point at token 0
    for (int i = base[e] + cnt[e] + tid; i < base[e] + pcnt[e]; i += 256)
      row2tok[i] = 0;
}

// ---------------------------------------------------------------------------
// K3/K4: tiled WMMA GEMM over routed rows, double-buffered LDS filled by
// CDNA5 async global->LDS DMA (ASYNCcnt), overlapping next-tile fill with
// current-tile WMMA compute.
// C[slot, n] = A[row(slot), :K] * Bt[e][n, :K] + bias[e][n]  (Bt is N-major)
// 256 threads = 8 waves in 4(M)x2(N) grid; each wave: 2x4 16x16 frags.
// ---------------------------------------------------------------------------
template <int KDIM, int NSIZE, bool GATHER, bool RELU, bool OUTBF>
__global__ __launch_bounds__(256, 2) void moe_gemm(
    const bf16* __restrict__ A, const bf16* __restrict__ Bt,
    const float* __restrict__ bias, const int* __restrict__ meta,
    const int* __restrict__ row2tok, bf16* __restrict__ outB,
    float* __restrict__ outF) {
  const int mt = blockIdx.x;
  if (mt >= meta[0]) return;             // unused padded tile
  const int e = meta[32 + mt];
  const int m0 = mt * TM;
  const int n0 = blockIdx.y * TN;

  __shared__ __align__(16) bf16 As[2][TM][LDSW];
  __shared__ __align__(16) bf16 Bs[2][TN][LDSW];

  const int tid = threadIdx.x;
  const int srow = tid >> 1;             // 0..127 staging row
  const int shalf = (tid & 1) * 32;      // 32 bf16 (64B) per thread

  int arow;
  if (GATHER) arow = row2tok[m0 + srow];
  else        arow = m0 + srow;
  const bf16* aSrc = A + (size_t)arow * KDIM + shalf;
  const bf16* bSrc = Bt + (size_t)e * NSIZE * KDIM +
                     (size_t)(n0 + srow) * KDIM + shalf;
  // LDS byte offsets of this thread's staging slots (generic ptr low 32 bits)
  const unsigned aDst0 = (unsigned)(size_t)&As[0][srow][shalf];
  const unsigned bDst0 = (unsigned)(size_t)&Bs[0][srow][shalf];
  const unsigned aBufStride = (unsigned)sizeof(As[0]);
  const unsigned bBufStride = (unsigned)sizeof(Bs[0]);

  const int lane = tid & 31;
  const int wave = tid >> 5;
  const int wm = wave & 3;               // 4 M-groups of 32
  const int wn = wave >> 2;              // 2 N-groups of 64
  const int lm = lane & 15;
  const bool hiHalf = lane >= 16;

  v8f acc[2][4];
#pragma unroll
  for (int i = 0; i < 2; ++i)
#pragma unroll
    for (int j = 0; j < 4; ++j) acc[i][j] = (v8f)0.0f;

  // prologue: async-fill buffer 0 with k0 = 0
  {
    const char* ag = (const char*)(aSrc);
    const char* bg = (const char*)(bSrc);
#pragma unroll
    for (int j = 0; j < 4; ++j) {
      async_ld_b128(aDst0 + j * 16, ag + j * 16);
      async_ld_b128(bDst0 + j * 16, bg + j * 16);
    }
  }

  for (int k0 = 0; k0 < KDIM; k0 += TKS) {
    const int buf = (k0 / TKS) & 1;
    wait_async0();                       // my DMA into `buf` is in LDS
    __syncthreads();                     // everyone's DMA done; buf^1 free
    if (k0 + TKS < KDIM) {               // kick next tile into buf^1 (overlaps
      const char* ag = (const char*)(aSrc + k0 + TKS);   //  with WMMA below)
      const char* bg = (const char*)(bSrc + k0 + TKS);
      const unsigned aD = aDst0 + (buf ^ 1) * aBufStride;
      const unsigned bD = bDst0 + (buf ^ 1) * bBufStride;
#pragma unroll
      for (int j = 0; j < 4; ++j) {
        async_ld_b128(aD + j * 16, ag + j * 16);
        async_ld_b128(bD + j * 16, bg + j * 16);
      }
    }
#pragma unroll
    for (int kk = 0; kk < TKS; kk += 32) {
      v16bf af[2], bfr[4];
#pragma unroll
      for (int mf = 0; mf < 2; ++mf) {   // A frag: lane L holds M=L%16,
        int m = wm * 32 + mf * 16 + lm;  // K {0..7,16..23} / {8..15,24..31}
        int ko = kk + (hiHalf ? 8 : 0);
        v8bf lo = *(const v8bf*)&As[buf][m][ko];
        v8bf hi = *(const v8bf*)&As[buf][m][ko + 16];
        af[mf] = concat8(lo, hi);
      }
#pragma unroll
      for (int nf = 0; nf < 4; ++nf) {   // B frag: lane L holds N=L%16,
        int n = wn * 64 + nf * 16 + lm;  // K {0..7,16..23} / {8..15,24..31}
        int ko = kk + (hiHalf ? 8 : 0);
        v8bf lo = *(const v8bf*)&Bs[buf][n][ko];
        v8bf hi = *(const v8bf*)&Bs[buf][n][ko + 16];
        bfr[nf] = concat8(lo, hi);
      }
#pragma unroll
      for (int mf = 0; mf < 2; ++mf)
#pragma unroll
        for (int nf = 0; nf < 4; ++nf)
          acc[mf][nf] = __builtin_amdgcn_wmma_f32_16x16x32_bf16(
              false, af[mf], false, bfr[nf], (short)0, acc[mf][nf],
              false, false);
    }
  }

  // epilogue: C/D layout -> lane L, vgpr r: M=r+(L>=16?8:0), N=L%16
  const float* be = bias + (size_t)e * NSIZE + n0;
  const int moff = hiHalf ? 8 : 0;
#pragma unroll
  for (int mf = 0; mf < 2; ++mf) {
#pragma unroll
    for (int nf = 0; nf < 4; ++nf) {
      int col = wn * 64 + nf * 16 + lm;
      float bv = be[col];
#pragma unroll
      for (int r = 0; r < 8; ++r) {
        int row = m0 + wm * 32 + mf * 16 + r + moff;
        float v = acc[mf][nf][r] + bv;
        if (RELU) v = fmaxf(v, 0.0f);
        size_t o = (size_t)row * NSIZE + n0 + col;
        if (OUTBF) outB[o] = f2bf(v);
        else       outF[o] = v;
      }
    }
  }
}

// ---------------------------------------------------------------------------
// K5: combine. out[t,:] = p0*y[slot0,:] + p1*y[slot1,:]
// ---------------------------------------------------------------------------
__global__ __launch_bounds__(256) void combine_kernel(
    const float* __restrict__ y, const int* __restrict__ tok_slot,
    const float* __restrict__ topk_prob, float* __restrict__ out) {
  const int t = blockIdx.x;
  const int c4 = threadIdx.x;            // 256 float4 per token (D=1024)
  const int s0 = tok_slot[t * 2 + 0], s1 = tok_slot[t * 2 + 1];
  const float p0 = topk_prob[t * 2 + 0], p1 = topk_prob[t * 2 + 1];
  float4 a = ((const float4*)(y + (size_t)s0 * DM))[c4];
  float4 b = ((const float4*)(y + (size_t)s1 * DM))[c4];
  float4 o;
  o.x = p0 * a.x + p1 * b.x;  o.y = p0 * a.y + p1 * b.y;
  o.z = p0 * a.z + p1 * b.z;  o.w = p0 * a.w + p1 * b.w;
  ((float4*)(out + (size_t)t * DM))[c4] = o;
}

// ---------------------------------------------------------------------------
extern "C" void kernel_launch(void* const* d_in, const int* in_sizes, int n_in,
                              void* d_out, int out_size, void* d_ws,
                              size_t ws_size, hipStream_t stream) {
  (void)in_sizes; (void)n_in; (void)out_size; (void)ws_size;
  const float* x      = (const float*)d_in[0];
  const float* gate_w = (const float*)d_in[1];
  const float* gate_b = (const float*)d_in[2];
  const float* w1     = (const float*)d_in[3];
  const float* b1     = (const float*)d_in[4];
  const float* w2     = (const float*)d_in[5];
  const float* b2     = (const float*)d_in[6];
  float* out = (float*)d_out;

  char* p = (char*)d_ws;
  auto alloc = [&](size_t bytes) -> void* {
    void* r = (void*)p;
    p += (bytes + 255) & ~(size_t)255;
    return r;
  };
  int*   meta      = (int*)alloc(256 * sizeof(int));
  int*   topk_idx  = (int*)alloc((size_t)T_TOK * 2 * sizeof(int));
  float* topk_prob = (float*)alloc((size_t)T_TOK * 2 * sizeof(float));
  int*   tok_slot  = (int*)alloc((size_t)T_TOK * 2 * sizeof(int));
  int*   row2tok   = (int*)alloc((size_t)MAX_ROWS * sizeof(int));
  bf16*  xb        = (bf16*)alloc((size_t)T_TOK * DM * sizeof(bf16));
  bf16*  w1t       = (bf16*)alloc((size_t)N_EXP * DM * DF * sizeof(bf16));
  bf16*  w2t       = (bf16*)alloc((size_t)N_EXP * DM * DF * sizeof(bf16));
  bf16*  h         = (bf16*)alloc((size_t)MAX_ROWS * DF * sizeof(bf16));
  float* y         = (float*)alloc((size_t)MAX_ROWS * DM * sizeof(float));

  // K0: weights f32 [E][K][N] -> bf16 N-major [E][N][K]
  transpose_convert<DM, DF><<<dim3(DF / 64, DM / 64, N_EXP), 256, 0, stream>>>(w1, w1t);
  transpose_convert<DF, DM><<<dim3(DM / 64, DF / 64, N_EXP), 256, 0, stream>>>(w2, w2t);
  // K1: gating (+ x -> bf16)
  gate_kernel<<<T_TOK / 8, 256, 0, stream>>>(x, gate_w, gate_b, xb, topk_idx, topk_prob);
  // K2: routing
  route_kernel<<<1, 256, 0, stream>>>(topk_idx, meta, tok_slot, row2tok);
  // K3: h = relu(gather(xb) @ w1[e] + b1[e])   [slots x DF] bf16
  moe_gemm<DM, DF, true, true, true>
      <<<dim3(MAX_MT, DF / TN), 256, 0, stream>>>(xb, w1t, b1, meta, row2tok, h, nullptr);
  // K4: y = h @ w2[e] + b2[e]                  [slots x DM] f32
  moe_gemm<DF, DM, false, false, false>
      <<<dim3(MAX_MT, DM / TN), 256, 0, stream>>>(h, w2t, b2, meta, row2tok, nullptr, y);
  // K5: weighted combine
  combine_kernel<<<T_TOK, 256, 0, stream>>>(y, tok_slot, topk_prob, out);
}
